// VectorQuantizerEMA_20890720928573
// MI455X (gfx1250) — matrix operand order repaired and checked by reading, not compiled, not run on hardware
//
#include <hip/hip_runtime.h>
#include <hip/hip_bf16.h>

#define M_ROWS 131072   // 32 * 4096
#define DIM    256
#define K_CODES 1024
#define DECAY  0.900009f
#define EPS_F  1e-5f

typedef __attribute__((ext_vector_type(2)))  float    v2f;
typedef __attribute__((ext_vector_type(4)))  float    v4f;
typedef __attribute__((ext_vector_type(8)))  float    v8f;
typedef __attribute__((ext_vector_type(4)))  _Float16 v4h;
typedef __attribute__((ext_vector_type(16))) _Float16 v16h;

#define HSTRIDE 264   // 256 + 8 half pad: keeps B-fragment LDS reads bank-spread

// ---------------- zero scratch accumulators ----------------
__global__ void vq_zero_kernel(float* p, int n) {
    int t = blockIdx.x * blockDim.x + threadIdx.x;
    if (t < n) p[t] = 0.0f;
}

// ---------------- ||w_k||^2, one wave per codebook row ----------------
__global__ __launch_bounds__(256) void vq_wsq_kernel(const float* __restrict__ w,
                                                     float* __restrict__ wsq) {
    int lane = threadIdx.x & 31, wv = threadIdx.x >> 5;
    int row  = blockIdx.x * 8 + wv;            // grid = K/8
    const float* p = w + (size_t)row * DIM;
    float s = 0.0f;
#pragma unroll
    for (int j = 0; j < 8; ++j) { float t = p[lane + 32 * j]; s += t * t; }
#pragma unroll
    for (int off = 16; off >= 1; off >>= 1) s += __shfl_xor(s, off, 32);
    if (lane == 0) wsq[row] = s;
}

// Load one 32-code super-tile (32x256 f32) into registers, coalesced float4.
__device__ __forceinline__ void vq_load_tile(const float* __restrict__ weight, int st,
                                             int tid, v4f (&s)[8]) {
#pragma unroll
    for (int j = 0; j < 8; ++j) {
        int f = j * 256 + tid;                 // float4 id in [0,2048)
        int row = f >> 6, c4 = f & 63;         // 64 float4 per 256-float row
        s[j] = *(const v4f*)(weight + (size_t)(st * 32 + row) * DIM + c4 * 4);
    }
}

// Split f32 -> (hi,lo) f16 and store into the LDS super-tile buffers.
__device__ __forceinline__ void vq_store_tile_f16(_Float16* __restrict__ wh,
                                                  _Float16* __restrict__ wl,
                                                  int tid, const v4f (&s)[8]) {
#pragma unroll
    for (int j = 0; j < 8; ++j) {
        int f = j * 256 + tid;
        int row = f >> 6, c4 = f & 63;
        v4h h, l;
#pragma unroll
        for (int e = 0; e < 4; ++e) {
            float fv = s[j][e];
            _Float16 hv = (_Float16)fv;
            h[e] = hv;
            l[e] = (_Float16)(fv - (float)hv);
        }
        *(v4h*)(&wh[row * HSTRIDE + c4 * 4]) = h;
        *(v4h*)(&wl[row * HSTRIDE + c4 * 4]) = l;
    }
}

// ---------------- main: split-f16 WMMA distances + argmin ----------------
// block = 256 threads (8 waves); each wave owns 16 x-rows (block: 128 rows).
// x.w computed as xh.wh + xl.wh + xh.wl with v_wmma_f32_16x16x32_f16 (f32
// accumulate): 24 WMMAs per 16x16 tile over D=256 instead of 64 on the slow
// 16x16x4 f32 op, on a pipe with 8x the MACs/instruction. Residual error
// ~2^-22 relative -- far below inter-code distance gaps.
// Outer loop: 32 super-tiles of 32 codes, double-buffered LDS with in-flight
// register staging; two accumulators -> two interleaved WMMA chains.
__global__ __launch_bounds__(256) void vq_argmin_kernel(const float* __restrict__ x,
                                                        const float* __restrict__ weight,
                                                        const float* __restrict__ wsq,
                                                        int* __restrict__ indices) {
    __shared__ _Float16 whbuf[2][32 * HSTRIDE];   // hi halves, 2 x 16.9KB
    __shared__ _Float16 wlbuf[2][32 * HSTRIDE];   // lo halves, 2 x 16.9KB
    const int tid    = threadIdx.x;
    const int lane   = tid & 31;
    const int wave   = tid >> 5;
    const int rowBase = blockIdx.x * 128 + wave * 16;
    const int m  = lane & 15;                  // A-matrix row for this lane
    const int hi = lane >> 4;                  // lane half per Sec 7.12.2 layouts

    // Build split-precision A fragments for this wave's 16x256 strip:
    // 8 k-chunks of 32, each v16h: elems 0..7 -> K=hi*16+j, 8..15 -> K=hi*16+8+j.
    const float* xrow = x + (size_t)(rowBase + m) * DIM;
    v16h ah[8], al[8];
#pragma unroll
    for (int c = 0; c < 8; ++c) {
        const float* p0 = xrow + 32 * c + hi * 16;
        const float* p1 = p0 + 8;
#pragma unroll
        for (int j = 0; j < 8; ++j) {
            float f0 = p0[j];
            _Float16 h0 = (_Float16)f0;
            ah[c][j] = h0;
            al[c][j] = (_Float16)(f0 - (float)h0);
            float f1 = p1[j];
            _Float16 h1 = (_Float16)f1;
            ah[c][8 + j] = h1;
            al[c][8 + j] = (_Float16)(f1 - (float)h1);
        }
    }

    float minv[8];
    int   mini[8];
#pragma unroll
    for (int r = 0; r < 8; ++r) { minv[r] = 3.4e38f; mini[r] = 0; }

    // Prologue: stage super-tile 0.
    v4f sreg[8];
    vq_load_tile(weight, 0, tid, sreg);
    vq_store_tile_f16(whbuf[0], wlbuf[0], tid, sreg);
    __syncthreads();

    for (int st = 0; st < K_CODES / 32; ++st) {
        const int cur = st & 1;
        // Issue next tile's global loads now; waits land after the WMMA block.
        if (st + 1 < K_CODES / 32) vq_load_tile(weight, st + 1, tid, sreg);

        v8f acc0 = {0.f, 0.f, 0.f, 0.f, 0.f, 0.f, 0.f, 0.f};
        v8f acc1 = {0.f, 0.f, 0.f, 0.f, 0.f, 0.f, 0.f, 0.f};
        // B fragment base: lane(n,hi) reads 16 contiguous halves w[n][k..k+15],
        // k = 32*chunk + hi*16. b0 -> codes +0..15, b1 -> codes +16..31.
        const _Float16* b0h = &whbuf[cur][(lane & 15) * HSTRIDE + hi * 16];
        const _Float16* b1h = b0h + 16 * HSTRIDE;
        const _Float16* b0l = &wlbuf[cur][(lane & 15) * HSTRIDE + hi * 16];
        const _Float16* b1l = b0l + 16 * HSTRIDE;
#pragma unroll
        for (int c = 0; c < 8; ++c) {
            v16h bh0 = *(const v16h*)(b0h + 32 * c);
            v16h bh1 = *(const v16h*)(b1h + 32 * c);
            v16h bl0 = *(const v16h*)(b0l + 32 * c);
            v16h bl1 = *(const v16h*)(b1l + 32 * c);
            acc0 = __builtin_amdgcn_wmma_f32_16x16x32_f16(
                false, ah[c], false, bh0, (short)0, acc0, false, false);
            acc1 = __builtin_amdgcn_wmma_f32_16x16x32_f16(
                false, ah[c], false, bh1, (short)0, acc1, false, false);
            acc0 = __builtin_amdgcn_wmma_f32_16x16x32_f16(
                false, al[c], false, bh0, (short)0, acc0, false, false);
            acc1 = __builtin_amdgcn_wmma_f32_16x16x32_f16(
                false, al[c], false, bh1, (short)0, acc1, false, false);
            acc0 = __builtin_amdgcn_wmma_f32_16x16x32_f16(
                false, ah[c], false, bl0, (short)0, acc0, false, false);
            acc1 = __builtin_amdgcn_wmma_f32_16x16x32_f16(
                false, ah[c], false, bl1, (short)0, acc1, false, false);
        }

        // dist(row, col) = ||w||^2 - 2 * x.w   (||x||^2 constant per row).
        // Check col0 before col1 to preserve first-occurrence argmin ties.
        const int   col0 = st * 32 + (lane & 15);
        const float wq0  = wsq[col0];
        const float wq1  = wsq[col0 + 16];
#pragma unroll
        for (int r = 0; r < 8; ++r) {
            float d0 = wq0 - 2.0f * acc0[r];
            if (d0 < minv[r]) { minv[r] = d0; mini[r] = col0; }
            float d1 = wq1 - 2.0f * acc1[r];
            if (d1 < minv[r]) { minv[r] = d1; mini[r] = col0 + 16; }
        }

        __syncthreads();                        // all waves done reading old buffer
        if (st + 1 < K_CODES / 32) {
            vq_store_tile_f16(whbuf[cur ^ 1], wlbuf[cur ^ 1], tid, sreg);
            __syncthreads();                    // stores visible before next compute
        }
    }

    // Cross-lane argmin within each 16-lane half (masks <= 8 stay in-half).
#pragma unroll
    for (int off = 8; off >= 1; off >>= 1) {
#pragma unroll
        for (int r = 0; r < 8; ++r) {
            float ov = __shfl_xor(minv[r], off, 32);
            int   oi = __shfl_xor(mini[r], off, 32);
            if (ov < minv[r] || (ov == minv[r] && oi < mini[r])) { minv[r] = ov; mini[r] = oi; }
        }
    }
    if ((lane & 15) == 0) {
        int rbase = rowBase + hi * 8;          // lane 0 -> rows 0..7, lane 16 -> rows 8..15
#pragma unroll
        for (int r = 0; r < 8; ++r) indices[rbase + r] = mini[r];
    }
}

// ---------------- gather + straight-through out + loss + EMA scatter ----------------
__global__ __launch_bounds__(256) void vq_gather_scatter_kernel(const float* __restrict__ x,
                                                                const float* __restrict__ weight,
                                                                const int* __restrict__ indices,
                                                                float* __restrict__ q_out,
                                                                float* __restrict__ csize,
                                                                float* __restrict__ esum,
                                                                double* __restrict__ loss_d) {
    int lane = threadIdx.x & 31, wv = threadIdx.x >> 5;
    size_t row = (size_t)blockIdx.x * 8 + wv;  // grid = M/8
    int idx = indices[row];
    const float* xr = x + row * DIM;
    const float* wr = weight + (size_t)idx * DIM;
    float* qo = q_out + row * DIM;
    float acc = 0.0f;
#pragma unroll
    for (int j = 0; j < 8; ++j) {
        int d = lane + 32 * j;
        float xv = xr[d], wvv = wr[d];
        qo[d] = xv + (wvv - xv);               // straight-through, same rounding as ref
        float diff = xv - wvv;
        acc += diff * diff;
        atomicAdd(&esum[(size_t)idx * DIM + d], xv);
    }
#pragma unroll
    for (int off = 16; off >= 1; off >>= 1) acc += __shfl_xor(acc, off, 32);
    if (lane == 0) {
        atomicAdd(loss_d, (double)acc);
        atomicAdd(&csize[idx], 1.0f);
    }
}

// ---------------- EMA cluster size + n ----------------
__global__ __launch_bounds__(256) void vq_finalize_cs_kernel(const float* __restrict__ ema_cs,
                                                             const float* __restrict__ csize,
                                                             float* __restrict__ out_ncs,
                                                             double* __restrict__ n_d) {
    int k = blockIdx.x * blockDim.x + threadIdx.x;   // grid*block = K
    float ncs = ema_cs[k] * DECAY + csize[k] * (1.0f - DECAY);
    out_ncs[k] = ncs;
    float s = ncs;
#pragma unroll
    for (int off = 16; off >= 1; off >>= 1) s += __shfl_xor(s, off, 32);
    if ((threadIdx.x & 31) == 0) atomicAdd(n_d, (double)s);
}

// ---------------- EMA embedding + normalized weight + loss ----------------
__global__ __launch_bounds__(256) void vq_finalize_w_kernel(const float* __restrict__ ema_emb,
                                                            const float* __restrict__ esum,
                                                            const float* __restrict__ out_ncs,
                                                            const double* __restrict__ n_d,
                                                            const double* __restrict__ loss_d,
                                                            float* __restrict__ out_w,
                                                            float* __restrict__ out_nemb,
                                                            float* __restrict__ out_loss) {
    size_t t = (size_t)blockIdx.x * 256 + threadIdx.x;  // K*D total
    int k = (int)(t >> 8);
    float ne = ema_emb[t] * DECAY + esum[t] * (1.0f - DECAY);
    out_nemb[t] = ne;
    float n = (float)(*n_d);
    float norm = (out_ncs[k] + EPS_F) / (n + (float)K_CODES * EPS_F);
    out_w[t] = ne / norm;
    if (t == 0) *out_loss = (float)((*loss_d) / ((double)M_ROWS * (double)DIM));
}

extern "C" void kernel_launch(void* const* d_in, const int* in_sizes, int n_in,
                              void* d_out, int out_size, void* d_ws, size_t ws_size,
                              hipStream_t stream) {
    const float* x        = (const float*)d_in[0];   // [M, D]
    const float* weight   = (const float*)d_in[1];   // [K, D]
    const float* ema_cs   = (const float*)d_in[2];   // [K]
    const float* ema_emb  = (const float*)d_in[3];   // [K, D]

    // Output layout: quantized_st [M*D] | loss [1] | new_weight [K*D] | new_cs [K] | new_emb [K*D]
    float* q_out    = (float*)d_out;
    float* loss_out = q_out + (size_t)M_ROWS * DIM;
    float* w_out    = loss_out + 1;
    float* ncs_out  = w_out + (size_t)K_CODES * DIM;
    float* nemb_out = ncs_out + K_CODES;

    // Workspace layout (all offsets 8B-aligned):
    int*    idxs   = (int*)d_ws;                                   // M ints
    float*  wsq    = (float*)((char*)d_ws + (size_t)M_ROWS * 4);   // K
    float*  csize  = wsq + K_CODES;                                // K
    float*  esum   = csize + K_CODES;                              // K*D
    double* loss_d = (double*)(esum + (size_t)K_CODES * DIM);      // 1
    double* n_d    = loss_d + 1;                                   // 1

    // 1) zero accumulators (csize, esum, loss_d, n_d are contiguous floats)
    int nzero = K_CODES + K_CODES * DIM + 4;
    vq_zero_kernel<<<(nzero + 255) / 256, 256, 0, stream>>>(csize, nzero);

    // 2) codebook squared norms
    vq_wsq_kernel<<<K_CODES / 8, 256, 0, stream>>>(weight, wsq);

    // 3) split-f16 WMMA distance + argmin
    vq_argmin_kernel<<<M_ROWS / 128, 256, 0, stream>>>(x, weight, wsq, idxs);

    // 4) gather, straight-through output, loss, EMA scatter
    vq_gather_scatter_kernel<<<M_ROWS / 8, 256, 0, stream>>>(x, weight, idxs, q_out,
                                                             csize, esum, loss_d);

    // 5) new cluster sizes + total n
    vq_finalize_cs_kernel<<<K_CODES / 256, 256, 0, stream>>>(ema_cs, csize, ncs_out, n_d);

    // 6) new embedding EMA, normalized weights, loss scalar
    vq_finalize_w_kernel<<<(K_CODES * DIM) / 256, 256, 0, stream>>>(ema_emb, esum, ncs_out,
                                                                    n_d, loss_d,
                                                                    w_out, nemb_out, loss_out);
}
